// DataEmbedding_ALLPE_Weighted_61744449848199
// MI455X (gfx1250) — matrix operand order repaired and checked by reading, not compiled, not run on hardware
//
#include <hip/hip_runtime.h>
#include <math.h>

// Problem constants (match reference)
#define BATCH 16
#define LSEQ  4096
#define CCH   32
#define DMODEL 512
#define WWIN  24
#define MTOK  (BATCH*LSEQ)   // 65536 tokens

typedef __bf16 bf16_t;
typedef bf16_t v16bf __attribute__((ext_vector_type(16)));
typedef float  v8f   __attribute__((ext_vector_type(8)));
typedef unsigned int u32x4 __attribute__((ext_vector_type(4)));
typedef unsigned int u32x2 __attribute__((ext_vector_type(2)));

#define AS3 __attribute__((address_space(3)))

__device__ __forceinline__ unsigned short f2bf(float f) {
  unsigned int u = __float_as_uint(f);
  u += 0x7FFFu + ((u >> 16) & 1u);        // round-to-nearest-even
  return (unsigned short)(u >> 16);
}

// ---------------------------------------------------------------------------
// Tensor Data Mover staging for the B panel (512 rows x 32 bf16, row pitch K).
// One instruction per K-step DMAs the whole 40KB panel into LDS with an 80B
// padded pitch (pad_interval = 16 dwords -> pad every 64B, pad_amount = 4
// dwords -> +16B), i.e. exactly the bank-conflict-free fragment layout.
// This toolchain's builtin is the 6-arg form:
//   (u32x4 g0, i32x8 g1, i32x4 g2, i32x4 g3, i32x8 extra, i32 cpol)
// ---------------------------------------------------------------------------
#if defined(__gfx1250__) && __has_builtin(__builtin_amdgcn_tensor_load_to_lds)
#define USE_TDM 1
typedef unsigned int tdu4 __attribute__((ext_vector_type(4)));
typedef int tdi8 __attribute__((ext_vector_type(8)));
typedef int tdi4 __attribute__((ext_vector_type(4)));

__device__ __forceinline__ void tdm_load_B(const unsigned short* Bm, int kk, int K,
                                           unsigned lds_off) {
  unsigned long long g = (unsigned long long)Bm + ((unsigned long long)(unsigned)kk << 1);
  tdu4 g0 = { 1u,                                          // count=1 (valid user D#)
              lds_off,                                     // lds_addr (bytes)
              (unsigned)(g & 0xFFFFFFFFull),               // global_addr[31:0]
              (unsigned)((g >> 32) & 0x01FFFFFFull) | (2u << 30) }; // addr[56:32]|type=2
  tdi8 g1 = { (1 << 16) | (1 << 20) | (3 << 22) | (3 << 25),
              // data_size=2B | pad_enable | pad_interval=16dw(64B) | pad_amount=4dw(16B)
              (int)((unsigned)(K & 0xFFFF) << 16),         // tensor_dim0[15:0] = K
              (int)(512u << 16),                           // dim0[31:16]=0 | tensor_dim1=512
              (int)(32u << 16),                            // dim1[31:16]=0 | tile_dim0=32
              512,                                         // tile_dim1=512 | tile_dim2=0
              K,                                           // tensor_dim0_stride[31:0]=K
              0, 0 };                                      // stride hi / dim1_stride = 0
  tdi4 gz4 = { 0, 0, 0, 0 };
  tdi8 gz8 = { 0, 0, 0, 0, 0, 0, 0, 0 };
  __builtin_amdgcn_tensor_load_to_lds(g0, g1, gz4, gz4, gz8, 0);
}
__device__ __forceinline__ void tdm_wait() {
#if __has_builtin(__builtin_amdgcn_s_wait_tensorcnt)
  __builtin_amdgcn_s_wait_tensorcnt((short)0);
#else
  asm volatile("s_wait_tensorcnt 0x0" ::: "memory");
#endif
}
#else
#define USE_TDM 0
#endif

// ---------------------------------------------------------------------------
// Kernel 1: rolling stats (mean/max/min/std over W=24) + lag diffs -> bf16
// comb[token][256] = [x, mean, max, min, std, lag3, lag5, lag7] each 32 ch
// ---------------------------------------------------------------------------
__global__ __launch_bounds__(256) void feat_kernel(const float* __restrict__ x,
                                                   unsigned short* __restrict__ comb) {
  int tok = blockIdx.x * 8 + (threadIdx.x >> 5);
  int c   = threadIdx.x & 31;
  int b = tok >> 12, l = tok & 4095;
  const float* xb = x + ((long)b * LSEQ) * CCH;
  float s = 0.f, s2 = 0.f, mx = -3.4e38f, mn = 3.4e38f;
  float v3 = 0.f, v5 = 0.f, v7 = 0.f, vx = 0.f;
  #pragma unroll
  for (int j = 0; j < WWIN; ++j) {
    int t = l - (WWIN - 1) + j; t = t < 0 ? 0 : t;      // replicate pad
    float v = xb[(long)t * CCH + c];
    s += v; s2 += v * v; mx = fmaxf(mx, v); mn = fminf(mn, v);
    if (j == 16) v7 = v;        // l-7
    if (j == 18) v5 = v;        // l-5
    if (j == 20) v3 = v;        // l-3
    if (j == 23) vx = v;        // l
  }
  float mean = s * (1.f / 24.f);
  float var  = fmaxf((s2 - 24.f * mean * mean) * (1.f / 23.f), 0.f); // unbiased
  float sd   = sqrtf(var);
  unsigned short* row = comb + (long)tok * 256;
  row[  0 + c] = f2bf(vx);      row[ 32 + c] = f2bf(mean);
  row[ 64 + c] = f2bf(mx);      row[ 96 + c] = f2bf(mn);
  row[128 + c] = f2bf(sd);      row[160 + c] = f2bf(vx - v3);
  row[192 + c] = f2bf(vx - v5); row[224 + c] = f2bf(vx - v7);
}

// ---------------------------------------------------------------------------
// Weight packing: all GEMM B matrices stored N-major [N][K] in bf16
// ---------------------------------------------------------------------------
__global__ void pack_conv(const float* __restrict__ cw, unsigned short* __restrict__ Bn1) {
  int id = blockIdx.x * 256 + threadIdx.x;          // 512*768
  int d = id / 768, rem = id % 768;
  int tap = rem >> 8, i = rem & 255;                // K index = tap*256 + ch
  Bn1[id] = f2bf(cw[(long)d * 768 + i * 3 + tap]);  // conv_w[d][i][tap]
}
__global__ void pack_mix(const float* __restrict__ mw, unsigned short* __restrict__ Bn3) {
  int id = blockIdx.x * 256 + threadIdx.x;          // 512*512
  int d = id >> 9, k = id & 511;
  Bn3[id] = f2bf(mw[(long)d * 1024 + 512 + k]);     // mixer_w[d][512+k]
}
__global__ void pack_tape(const float* __restrict__ tp, unsigned short* __restrict__ At) {
  int id = blockIdx.x * 256 + threadIdx.x;          // 4096*512 (first L rows)
  At[id] = f2bf(tp[id]);
}

// Fused weight Wf[o][j] = sum_k mixer_w[o][k] * tproj_w[k][j]  (bf16, N-major)
__global__ __launch_bounds__(256) void wf_kernel(const float* __restrict__ mw,
                                                 const float* __restrict__ tw,
                                                 unsigned short* __restrict__ Bn2) {
  __shared__ float tA[16][17], tB[16][17];
  int tx = threadIdx.x & 15, ty = threadIdx.x >> 4;
  int o0 = blockIdx.y * 16, j0 = blockIdx.x * 16;
  float acc = 0.f;
  for (int kk = 0; kk < 512; kk += 16) {
    tA[ty][tx] = mw[(long)(o0 + ty) * 1024 + kk + tx];
    tB[ty][tx] = tw[(long)(kk + ty) * 512 + j0 + tx];
    __syncthreads();
    #pragma unroll
    for (int e = 0; e < 16; ++e) acc += tA[ty][e] * tB[e][tx];
    __syncthreads();
  }
  Bn2[(long)(o0 + ty) * 512 + j0 + tx] = f2bf(acc);
}

// bias2[o] = mixer_b[o] + sum_k tproj_b[k] * mixer_w[o][k]
__global__ void bias2_kernel(const float* __restrict__ mb, const float* __restrict__ mw,
                             const float* __restrict__ tb, float* __restrict__ bias2) {
  int o = blockIdx.x * 256 + threadIdx.x;
  float a = mb[o];
  for (int k = 0; k < 512; ++k) a += tb[k] * mw[(long)o * 1024 + k];
  bias2[o] = a;
}

// ---------------------------------------------------------------------------
// base[l][d] = w1*LN(pe_fixed)[d] + w2*LN(learned_pe[l])[d]
// ---------------------------------------------------------------------------
__global__ __launch_bounds__(256) void base_kernel(const float* __restrict__ lpe,
    const float* __restrict__ gf, const float* __restrict__ bfv,
    const float* __restrict__ gl, const float* __restrict__ bl,
    const float* __restrict__ wp, float* __restrict__ base) {
  __shared__ float red[256];
  __shared__ float st[4];   // mean_f, rstd_f, mean_l, rstd_l
  __shared__ float sw[4];
  int l = blockIdx.x, t = threadIdx.x;
  float pf[2], pl[2];
  #pragma unroll
  for (int h = 0; h < 2; ++h) {
    int d = t + h * 256;
    int de = d & ~1;
    float div = expf((float)de * (-9.210340371976184f / 512.f)); // -ln(1e4)/D
    float ang = (float)l * div;
    pf[h] = (d & 1) ? cosf(ang) : sinf(ang);
    pl[h] = lpe[(long)l * 512 + d];
  }
  for (int which = 0; which < 2; ++which) {
    float a0 = which ? pl[0] : pf[0];
    float a1 = which ? pl[1] : pf[1];
    red[t] = a0 + a1; __syncthreads();
    for (int o = 128; o; o >>= 1) { if (t < o) red[t] += red[t + o]; __syncthreads(); }
    float m = red[0] * (1.f / 512.f); __syncthreads();
    float d0 = a0 - m, d1 = a1 - m;
    red[t] = d0 * d0 + d1 * d1; __syncthreads();
    for (int o = 128; o; o >>= 1) { if (t < o) red[t] += red[t + o]; __syncthreads(); }
    if (t == 0) { st[which * 2] = m; st[which * 2 + 1] = rsqrtf(red[0] * (1.f / 512.f) + 1e-5f); }
    __syncthreads();
  }
  if (t == 0) {
    float a = wp[0], b = wp[1], c = wp[2], d = wp[3];
    float mx = fmaxf(fmaxf(a, b), fmaxf(c, d));
    float e0 = expf(a - mx), e1 = expf(b - mx), e2 = expf(c - mx), e3 = expf(d - mx);
    float inv = 1.f / (e0 + e1 + e2 + e3);
    sw[0] = e0 * inv; sw[1] = e1 * inv; sw[2] = e2 * inv; sw[3] = e3 * inv;
  }
  __syncthreads();
  #pragma unroll
  for (int h = 0; h < 2; ++h) {
    int d = t + h * 256;
    float vf = gf[d] * (pf[h] - st[0]) * st[1] + bfv[d];
    float vl = gl[d] * (pl[h] - st[2]) * st[3] + bl[d];
    base[(long)l * 512 + d] = sw[1] * vf + sw[2] * vl;
  }
}

// ---------------------------------------------------------------------------
// Tiled bf16 WMMA GEMM, BM=64 x BN=512(full row) x BK=32, fused LN epilogue.
// LDS double-buffered; B panel staged by the Tensor Data Mover (wave 0 issues
// one tensor_load_to_lds per K-step, waits TENSORcnt, one barrier per step);
// A tile (4KB, circular rows for conv mode) register-pipelined per thread.
// MODE 0: pe_pre = tape@Wm2 + bias2                       (store f32)
// MODE 1: ce = LN(comb(3-tap circ)@convW + conv_b); out = w0*ce + base
// MODE 2: out += w3 * LN(ce@Wf + pe_pre)
// ---------------------------------------------------------------------------
template<int MODE>
__global__ __launch_bounds__(256) void gemm_ln(
    const unsigned short* __restrict__ A,
    const unsigned short* __restrict__ Bm,
    int K,
    const float* __restrict__ biasv,   // MODE0: bias2, MODE1: conv_b
    const float* __restrict__ gvec,
    const float* __restrict__ bvec,
    const float* __restrict__ addv,    // MODE1: base, MODE2: pe_pre
    const float* __restrict__ wp,
    unsigned short* __restrict__ ceOut,
    float* __restrict__ out) {
  __shared__ __align__(16) unsigned short sA[2][64 * 40];    // 2 x  5 KB
  __shared__ __align__(16) unsigned short sB[2][512 * 40];   // 2 x 40 KB
  __shared__ float redS[64], redS2[64], rMean[64], rRstd[64];
  __shared__ float sw[4];

  const int mBase = blockIdx.x * 64;
  const int tid = threadIdx.x;
  const int wv = tid >> 5, lane = tid & 31;
  const int ln15 = lane & 15, kh = lane >> 4;

  if (MODE != 0 && tid == 0) {
    float a = wp[0], b = wp[1], c = wp[2], d = wp[3];
    float mx = fmaxf(fmaxf(a, b), fmaxf(c, d));
    float e0 = expf(a - mx), e1 = expf(b - mx), e2 = expf(c - mx), e3 = expf(d - mx);
    float inv = 1.f / (e0 + e1 + e2 + e3);
    sw[0] = e0 * inv; sw[1] = e1 * inv; sw[2] = e2 * inv; sw[3] = e3 * inv;
  }

  const int ar = tid >> 2, ac = tid & 3;   // A stage: 64 rows x 4 thr (8 bf16 each)

  auto loadA = [&](int kk) -> u32x4 {
    long idx;
    if (MODE == 1) {            // implicit im2col: tap from K offset, circular row
      int m = mBase + ar;
      int bidx = m >> 12, l = m & 4095;
      int tap = kk >> 8;                       // 0,1,2 -> l-1, l, l+1 (mod L)
      int ls = (l + tap + 4095) & 4095;
      idx = (((long)bidx << 12) | ls) * 256 + (kk & 255) + ac * 8;
    } else {
      idx = (long)(mBase + ar) * K + kk + ac * 8;
    }
    return *(const u32x4*)(A + idx);
  };
#if !USE_TDM
  auto loadB = [&](int i, int kk) -> u32x4 {
    int flat = tid + i * 256;
    int bn = flat >> 2, bc = flat & 3;         // 512 rows x 4 chunks(16B)
    return *(const u32x4*)(Bm + (long)bn * K + kk + bc * 8);
  };
#endif

  v8f acc[4][4];
  {
    v8f z = {0.f, 0.f, 0.f, 0.f, 0.f, 0.f, 0.f, 0.f};
    #pragma unroll
    for (int i = 0; i < 4; ++i)
      #pragma unroll
      for (int j = 0; j < 4; ++j) acc[i][j] = z;
  }

#if USE_TDM
  const unsigned offB0 = (unsigned)(unsigned long long)(AS3 unsigned short*)&sB[0][0];
  const unsigned offB1 = (unsigned)(unsigned long long)(AS3 unsigned short*)&sB[1][0];
#endif

  // ---- prologue: stage step 0 into buffer 0 ----
  {
    u32x4 ra = loadA(0);
    *(u32x4*)&sA[0][ar * 40 + ac * 8] = ra;
#if USE_TDM
    if (wv == 0) { tdm_load_B(Bm, 0, K, offB0); tdm_wait(); }
#else
    u32x4 rb[8];
    #pragma unroll
    for (int i = 0; i < 8; ++i) rb[i] = loadB(i, 0);
    #pragma unroll
    for (int i = 0; i < 8; ++i) {
      int flat = tid + i * 256;
      *(u32x4*)&sB[0][(flat >> 2) * 40 + (flat & 3) * 8] = rb[i];
    }
#endif
  }
  __syncthreads();

  const int nsteps = K >> 5;
  union AF { v16bf v; u32x4 q[2]; };

  for (int s = 0; s < nsteps; ++s) {
    const int cur = s & 1, nxt = cur ^ 1;
    const bool more = (s + 1 < nsteps);
#if USE_TDM
    if (more && wv == 0) tdm_load_B(Bm, (s + 1) << 5, K, nxt ? offB1 : offB0);
#endif
    u32x4 ra2 = {0, 0, 0, 0};
    if (more) ra2 = loadA((s + 1) << 5);
#if !USE_TDM
    u32x4 rb2[8];
    if (more) {
      #pragma unroll
      for (int i = 0; i < 8; ++i) rb2[i] = loadB(i, (s + 1) << 5);
    }
    if (s + 2 < nsteps) {       // global_prefetch_b8 on the B stream (L2 warm)
      int bn = tid >> 2, bc = tid & 3;
      __builtin_prefetch(Bm + (long)bn * K + ((s + 2) << 5) + bc * 8, 0, 1);
    }
#endif
    // A fragments (ISA 16-bit A 16x32 layout: q0->VGPR0-3, q1->VGPR4-7)
    AF a[4];
    #pragma unroll
    for (int mt = 0; mt < 4; ++mt) {
      const unsigned short* p = &sA[cur][(mt * 16 + ln15) * 40 + kh * 8];
      a[mt].q[0] = *(const u32x4*)p;
      a[mt].q[1] = *(const u32x4*)(p + 16);
    }
    #pragma unroll
    for (int j = 0; j < 4; ++j) {
      AF bf;                    // B 32x16: lanes0-15 K0-7,16-23, lanes16-31 K8-15,24-31
      const unsigned short* p = &sB[cur][((wv * 4 + j) * 16 + ln15) * 40 + kh * 8];
      bf.q[0] = *(const u32x4*)p;
      bf.q[1] = *(const u32x4*)(p + 16);
      #pragma unroll
      for (int mt = 0; mt < 4; ++mt)
        acc[mt][j] = __builtin_amdgcn_wmma_f32_16x16x32_bf16(
            false, a[mt].v, false, bf.v, (short)0, acc[mt][j], false, false);
    }
    if (more) {
      *(u32x4*)&sA[nxt][ar * 40 + ac * 8] = ra2;
#if !USE_TDM
      #pragma unroll
      for (int i = 0; i < 8; ++i) {
        int flat = tid + i * 256;
        *(u32x4*)&sB[nxt][(flat >> 2) * 40 + (flat & 3) * 8] = rb2[i];
      }
#endif
    }
#if USE_TDM
    if (more && wv == 0) tdm_wait();
#endif
    __syncthreads();
  }

  // ------------------------- epilogue -------------------------
  int ncol[4]; float gv[4], bv[4], biasj[4];
  #pragma unroll
  for (int j = 0; j < 4; ++j) {
    ncol[j] = (wv * 4 + j) * 16 + ln15;
    if (MODE == 0 || MODE == 1) biasj[j] = biasv[ncol[j]];
    if (MODE != 0) { gv[j] = gvec[ncol[j]]; bv[j] = bvec[ncol[j]]; }
  }

  if (MODE == 0) {              // plain store (pe_pre), no LN
    #pragma unroll
    for (int mt = 0; mt < 4; ++mt)
      #pragma unroll
      for (int r = 0; r < 8; ++r) {
        long rg = mBase + mt * 16 + kh * 8 + r;
        #pragma unroll
        for (int j = 0; j < 4; ++j)
          out[rg * 512 + ncol[j]] = acc[mt][j][r] + biasj[j];
      }
    return;
  }

  if (tid < 64) { redS[tid] = 0.f; redS2[tid] = 0.f; }
  __syncthreads();

  #pragma unroll
  for (int mt = 0; mt < 4; ++mt) {
    #pragma unroll
    for (int r = 0; r < 8; ++r) {
      int row = mt * 16 + kh * 8 + r;
      long rg = mBase + row;
      int l = (int)(rg & 4095);
      float p1 = 0.f, p2 = 0.f;
      #pragma unroll
      for (int j = 0; j < 4; ++j) {
        float v = acc[mt][j][r];
        if (MODE == 1) v += biasj[j];
        else           v += addv[(long)l * 512 + ncol[j]];   // + pe_pre
        acc[mt][j][r] = v;
        p1 += v; p2 += v * v;
      }
      #pragma unroll
      for (int o = 8; o; o >>= 1) { p1 += __shfl_xor(p1, o, 16); p2 += __shfl_xor(p2, o, 16); }
      if (ln15 == 0) { atomicAdd(&redS[row], p1); atomicAdd(&redS2[row], p2); }
    }
  }
  __syncthreads();
  if (tid < 64) {
    float m = redS[tid] * (1.f / 512.f);
    float var = redS2[tid] * (1.f / 512.f) - m * m;
    rMean[tid] = m; rRstd[tid] = rsqrtf(fmaxf(var, 0.f) + 1e-5f);
  }
  __syncthreads();

  #pragma unroll
  for (int mt = 0; mt < 4; ++mt)
    #pragma unroll
    for (int r = 0; r < 8; ++r) {
      int row = mt * 16 + kh * 8 + r;
      long rg = mBase + row;
      int l = (int)(rg & 4095);
      float m = rMean[row], rs = rRstd[row];
      #pragma unroll
      for (int j = 0; j < 4; ++j) {
        float val = (acc[mt][j][r] - m) * rs * gv[j] + bv[j];
        long oi = rg * 512 + ncol[j];
        if (MODE == 1) {
          ceOut[oi] = f2bf(val);
          out[oi] = sw[0] * val + addv[(long)l * 512 + ncol[j]];  // w0*ce + base
        } else {
          out[oi] += sw[3] * val;                                 // += w3*pe_tape
        }
      }
    }
}

// ---------------------------------------------------------------------------
extern "C" void kernel_launch(void* const* d_in, const int* in_sizes, int n_in,
                              void* d_out, int out_size, void* d_ws, size_t ws_size,
                              hipStream_t stream) {
  const float* x      = (const float*)d_in[0];
  const float* conv_w = (const float*)d_in[1];
  const float* conv_b = (const float*)d_in[2];
  const float* lpe    = (const float*)d_in[3];
  const float* tpos   = (const float*)d_in[4];
  const float* tprojw = (const float*)d_in[5];
  const float* tprojb = (const float*)d_in[6];
  const float* mixw   = (const float*)d_in[7];
  const float* mixb   = (const float*)d_in[8];
  const float* g_c = (const float*)d_in[9],  *b_c = (const float*)d_in[10];
  const float* g_f = (const float*)d_in[11], *b_f = (const float*)d_in[12];
  const float* g_l = (const float*)d_in[13], *b_l = (const float*)d_in[14];
  const float* g_t = (const float*)d_in[15], *b_t = (const float*)d_in[16];
  const float* wp  = (const float*)d_in[17];
  float* out = (float*)d_out;

  char* w = (char*)d_ws;
  auto alloc = [&](size_t bytes) -> char* {
    char* p = w; w += (bytes + 255) & ~(size_t)255; return p;
  };
  unsigned short* comb = (unsigned short*)alloc((size_t)MTOK * 256 * 2);  // 33.6 MB
  unsigned short* Bn1  = (unsigned short*)alloc((size_t)512 * 768 * 2);
  unsigned short* Bn2  = (unsigned short*)alloc((size_t)512 * 512 * 2);
  unsigned short* Bn3  = (unsigned short*)alloc((size_t)512 * 512 * 2);
  unsigned short* Atap = (unsigned short*)alloc((size_t)4096 * 512 * 2);
  float* bias2 = (float*)alloc((size_t)512 * 4);
  float* baseB = (float*)alloc((size_t)4096 * 512 * 4);
  float* pepre = (float*)alloc((size_t)4096 * 512 * 4);
  unsigned short* ce = (unsigned short*)alloc((size_t)MTOK * 512 * 2);    // 67 MB

  feat_kernel<<<MTOK / 8, 256, 0, stream>>>(x, comb);
  pack_conv<<<(512 * 768) / 256, 256, 0, stream>>>(conv_w, Bn1);
  pack_mix<<<(512 * 512) / 256, 256, 0, stream>>>(mixw, Bn3);
  pack_tape<<<(4096 * 512) / 256, 256, 0, stream>>>(tpos, Atap);
  wf_kernel<<<dim3(32, 32), 256, 0, stream>>>(mixw, tprojw, Bn2);
  bias2_kernel<<<2, 256, 0, stream>>>(mixb, mixw, tprojb, bias2);
  base_kernel<<<4096, 256, 0, stream>>>(lpe, g_f, b_f, g_l, b_l, wp, baseB);

  // pe_pre[l][d] = tape_pos[l] @ mixer_w2 + bias2
  gemm_ln<0><<<4096 / 64, 256, 0, stream>>>(Atap, Bn3, 512, bias2,
      nullptr, nullptr, nullptr, wp, nullptr, pepre);
  // combined_emb (conv-as-GEMM, K=768) + LN; out = w0*ce + base
  gemm_ln<1><<<MTOK / 64, 256, 0, stream>>>(comb, Bn1, 768, conv_b,
      g_c, b_c, baseB, wp, ce, out);
  // out += w3 * LN(ce @ Wf + pe_pre)
  gemm_ln<2><<<MTOK / 64, 256, 0, stream>>>(ce, Bn2, 512, nullptr,
      g_t, b_t, pepre, wp, nullptr, out);
}
